// MoELayer_15006615734191
// MI455X (gfx1250) — matrix operand order repaired and checked by reading, not compiled, not run on hardware
//
#include <hip/hip_runtime.h>
#include <cstdint>
#include <cstddef>

#define NTOK 8192
#define DIM  512
#define HID  2048
#define NE   8

typedef __bf16 bf16;
typedef __attribute__((ext_vector_type(16))) __bf16 v16bf;
typedef __attribute__((ext_vector_type(8)))  float  v8f;
typedef __attribute__((ext_vector_type(4)))  float  f4v;

union FragAB { v16bf v; f4v f4[2]; };

__device__ inline v8f v8f_zero() {
  v8f z;
#pragma unroll
  for (int i = 0; i < 8; ++i) z[i] = 0.f;
  return z;
}

// gfx1250 async global->LDS copy (ASYNCcnt) + drain. LDS dest is a byte
// offset within the wave's LDS allocation (dynamic LDS starts at 0 here:
// the expert kernel declares no static __shared__).
__device__ inline void async_load_b128(unsigned lds_off, const float* gaddr) {
  asm volatile("global_load_async_to_lds_b128 %0, %1, off"
               :: "v"(lds_off), "v"(gaddr) : "memory");
}
__device__ inline void wait_async0() {
  asm volatile("s_wait_asynccnt 0" ::: "memory");
}

// ---------------------------------------------------------------------------
// Kernel 1: convert + transpose weights to bf16, K-contiguous for B-fragments.
//   W1t[e][h][d] = bf16(W1[e][d][h])   (GEMM1 B: k=d contiguous per h column)
//   W2t[e][d][h] = bf16(W2[e][h][d])   (GEMM2 B: k=h contiguous per d column)
// Also zeroes the per-expert slot counters.
// ---------------------------------------------------------------------------
__global__ void prep_kernel(const float* __restrict__ W1, const float* __restrict__ W2,
                            bf16* __restrict__ W1t, bf16* __restrict__ W2t,
                            int* __restrict__ cnt) {
  int idx = blockIdx.x * 256 + threadIdx.x;       // 0 .. 8*2048*512-1 (2^23)
  if (idx < NE) cnt[idx] = 0;
  {
    int d = idx & (DIM - 1);
    int h = (idx >> 9) & (HID - 1);
    int e = idx >> 20;
    W1t[idx] = (bf16)W1[((size_t)(e * DIM + d)) * HID + h];
  }
  {
    int h = idx & (HID - 1);
    int d = (idx >> 11) & (DIM - 1);
    int e = idx >> 20;
    W2t[idx] = (bf16)W2[((size_t)(e * HID + h)) * DIM + d];
  }
}

// ---------------------------------------------------------------------------
// Kernel 2: gating. One wave32 per token.
// ---------------------------------------------------------------------------
__global__ void gate_kernel(const float* __restrict__ x, const float* __restrict__ gW,
                            const float* __restrict__ gB, int* __restrict__ cnt,
                            int* __restrict__ tlist, float* __restrict__ tw,
                            float* __restrict__ imp_part) {
  __shared__ float probs_s[8][8];
  int tid = threadIdx.x;
  int wave = tid >> 5, lane = tid & 31;
  int n = blockIdx.x * 8 + wave;

  float acc[8];
#pragma unroll
  for (int e = 0; e < 8; ++e) acc[e] = 0.f;
  const float* xr = x + (size_t)n * DIM;
  for (int d = lane; d < DIM; d += 32) {
    float xv = xr[d];
    const f4v* w4 = reinterpret_cast<const f4v*>(gW + d * 8);
    f4v w0 = w4[0], w1 = w4[1];
    acc[0] += xv * w0[0]; acc[1] += xv * w0[1]; acc[2] += xv * w0[2]; acc[3] += xv * w0[3];
    acc[4] += xv * w1[0]; acc[5] += xv * w1[1]; acc[6] += xv * w1[2]; acc[7] += xv * w1[3];
  }
#pragma unroll
  for (int m = 16; m >= 1; m >>= 1) {
#pragma unroll
    for (int e = 0; e < 8; ++e) acc[e] += __shfl_xor(acc[e], m, 32);
  }

  if (lane == 0) {
    float l[8], mx = -1e30f;
#pragma unroll
    for (int e = 0; e < 8; ++e) { l[e] = acc[e] + gB[e]; mx = fmaxf(mx, l[e]); }
    float p[8], s = 0.f;
#pragma unroll
    for (int e = 0; e < 8; ++e) { p[e] = expf(l[e] - mx); s += p[e]; }
    float inv = 1.f / s;
#pragma unroll
    for (int e = 0; e < 8; ++e) { p[e] *= inv; probs_s[wave][e] = p[e]; }
    int i1 = 0;
#pragma unroll
    for (int e = 1; e < 8; ++e) if (p[e] > p[i1]) i1 = e;
    int i2 = (i1 == 0) ? 1 : 0;
#pragma unroll
    for (int e = 0; e < 8; ++e) if (e != i1 && p[e] > p[i2]) i2 = e;
    float ws = p[i1] + p[i2];
    tw[2 * n + 0] = p[i1] / ws;
    tw[2 * n + 1] = p[i2] / ws;
    int pos = atomicAdd(&cnt[i1], 1);
    tlist[i1 * NTOK + pos] = 2 * n + 0;
    pos = atomicAdd(&cnt[i2], 1);
    tlist[i2 * NTOK + pos] = 2 * n + 1;
  }
  __syncthreads();
  if (tid < 8) {
    float s = 0.f;
#pragma unroll
    for (int w = 0; w < 8; ++w) s += probs_s[w][tid];
    imp_part[blockIdx.x * 8 + tid] = s;
  }
}

// ---------------------------------------------------------------------------
// Kernel 3: dispatched expert MLP. Block = 32 token-slots of one expert,
// 512 threads = 16 waves, bf16 WMMA 16x16x32 / fp32 accumulate.
//   Each wave: 2 M-tiles x 8 h-tiles (GEMM1, K=512) -> every B-fragment
//   feeds two WMMAs (halves L2 weight traffic vs a 16-token tile).
//   X tile staged via global_load_async_to_lds_b128 (ASYNCcnt), converted
//   to bf16 once, then fused bias + LayerNorm(H) + exact GELU, and
//   GEMM2 (2 M-tiles x 2 d-tiles per wave, K=2048) scatters y+b2 into the
//   per-slot Ybuf (slot-unique rows -> no atomics, deterministic).
// ---------------------------------------------------------------------------
__global__ __launch_bounds__(512) void expert_kernel(
    const float* __restrict__ x, const float* __restrict__ b1,
    const float* __restrict__ lng, const float* __restrict__ lnb,
    const float* __restrict__ b2, const int* __restrict__ cnt,
    const int* __restrict__ tlist, const bf16* __restrict__ W1t,
    const bf16* __restrict__ W2t, float* __restrict__ Ybuf) {
  constexpr int MT   = 32;        // tokens per block (2 WMMA M-tiles)
  constexpr int XSTR = DIM + 8;   // bf16 units, 16B-aligned rows, bank-staggered
  constexpr int ASTR = HID + 8;
  constexpr unsigned XS_BYTES = (unsigned)MT * XSTR * 2u;
  extern __shared__ char smem[];
  bf16*  Xs    = (bf16*)smem;                                    // MT*XSTR bf16
  bf16*  As    = (bf16*)(smem + XS_BYTES);                       // MT*ASTR bf16
  float* AsTmp = (float*)(smem + XS_BYTES);                      // fp32 staging (reuses As)
  float* red   = (float*)(smem + XS_BYTES + (size_t)MT * ASTR * 2); // 16*MT*2
  float* stats = red + 16 * MT * 2;                              // mu[MT], rstd[MT]
  int*   ids   = (int*)(stats + 2 * MT);                         // MT

  int e  = blockIdx.y;
  int t0 = blockIdx.x * MT;
  int count = cnt[e];
  if (t0 >= count) return;
  int valid = count - t0; if (valid > MT) valid = MT;
  int tid = threadIdx.x;

  if (tid < MT) ids[tid] = (tid < valid) ? tlist[e * NTOK + t0 + tid] : -1;
  __syncthreads();

  // ---- async-stage 32 x-rows (fp32) into LDS, then convert to bf16 ----
  int row = tid >> 4;            // 0..31
  int c0  = (tid & 15) * 32;     // 32 floats per thread
  {
    int id = ids[row];
    if (id >= 0) {
      const float* xr = x + (size_t)(id >> 1) * DIM + c0;
      unsigned lds0 = XS_BYTES + (unsigned)(row * DIM + c0) * 4u;
#pragma unroll
      for (int j = 0; j < 32; j += 4) async_load_b128(lds0 + j * 4u, xr + j);
    }
  }
  wait_async0();
  __syncthreads();
  {
    int id = ids[row];
    if (id >= 0) {
#pragma unroll
      for (int i = 0; i < 32; ++i)
        Xs[row * XSTR + c0 + i] = (bf16)AsTmp[row * DIM + c0 + i];
    } else {
#pragma unroll
      for (int i = 0; i < 32; ++i) Xs[row * XSTR + c0 + i] = (bf16)0.f;
    }
  }
  __syncthreads();

  int wave = tid >> 5, lane = tid & 31;
  int sub = lane >> 4, nl = lane & 15;   // sub = half (lanes 16-31), nl = n/m index

  // ---------------- GEMM1: h = Xs @ W1[e] ----------------
  v8f acc0[8], acc1[8];
#pragma unroll
  for (int t = 0; t < 8; ++t) { acc0[t] = v8f_zero(); acc1[t] = v8f_zero(); }

  const bf16* XsA0 = Xs + nl * XSTR;          // M-tile 0: rows 0..15
  const bf16* XsA1 = Xs + (16 + nl) * XSTR;   // M-tile 1: rows 16..31
  for (int kk = 0; kk < DIM; kk += 32) {
    FragAB a0, a1;   // A 16x32 bf16: element i -> K = kk + i + (i&8) + sub*8
    a0.f4[0] = *reinterpret_cast<const f4v*>(XsA0 + kk + sub * 8);
    a0.f4[1] = *reinterpret_cast<const f4v*>(XsA0 + kk + 16 + sub * 8);
    a1.f4[0] = *reinterpret_cast<const f4v*>(XsA1 + kk + sub * 8);
    a1.f4[1] = *reinterpret_cast<const f4v*>(XsA1 + kk + 16 + sub * 8);
#pragma unroll
    for (int t = 0; t < 8; ++t) {
      int h = wave * 128 + t * 16 + nl;
      FragAB b;  // B 32x16 bf16: element i -> K = kk + i + (i&8) + sub*8
      const f4v* bp = reinterpret_cast<const f4v*>(
          W1t + ((size_t)(e * HID + h)) * DIM + kk + sub * 8);
      b.f4[0] = bp[0]; b.f4[1] = bp[2];
      acc0[t] = __builtin_amdgcn_wmma_f32_16x16x32_bf16(
          false, a0.v, false, b.v, (short)0, acc0[t], false, false);
      acc1[t] = __builtin_amdgcn_wmma_f32_16x16x32_bf16(
          false, a1.v, false, b.v, (short)0, acc1[t], false, false);
    }
  }

  // bias b1, then LayerNorm stats over H per token (row m = mt*16 + sub*8 + r)
#pragma unroll
  for (int t = 0; t < 8; ++t) {
    int h = wave * 128 + t * 16 + nl;
    float b1v = b1[e * HID + h];
#pragma unroll
    for (int r = 0; r < 8; ++r) { acc0[t][r] += b1v; acc1[t][r] += b1v; }
  }
  float s0[8], q0[8], s1[8], q1[8];
#pragma unroll
  for (int r = 0; r < 8; ++r) { s0[r] = q0[r] = s1[r] = q1[r] = 0.f; }
#pragma unroll
  for (int t = 0; t < 8; ++t) {
#pragma unroll
    for (int r = 0; r < 8; ++r) {
      float v0 = acc0[t][r]; s0[r] += v0; q0[r] += v0 * v0;
      float v1 = acc1[t][r]; s1[r] += v1; q1[r] += v1 * v1;
    }
  }
#pragma unroll
  for (int mask = 8; mask >= 1; mask >>= 1) {   // reduce within the 16-lane half
#pragma unroll
    for (int r = 0; r < 8; ++r) {
      s0[r] += __shfl_xor(s0[r], mask, 32); q0[r] += __shfl_xor(q0[r], mask, 32);
      s1[r] += __shfl_xor(s1[r], mask, 32); q1[r] += __shfl_xor(q1[r], mask, 32);
    }
  }
  if (nl == 0) {
#pragma unroll
    for (int r = 0; r < 8; ++r) {
      int m0 = sub * 8 + r, m1 = 16 + sub * 8 + r;
      red[(wave * MT + m0) * 2 + 0] = s0[r];
      red[(wave * MT + m0) * 2 + 1] = q0[r];
      red[(wave * MT + m1) * 2 + 0] = s1[r];
      red[(wave * MT + m1) * 2 + 1] = q1[r];
    }
  }
  __syncthreads();
  if (tid < MT) {
    float S = 0.f, Q = 0.f;
#pragma unroll
    for (int w = 0; w < 16; ++w) {
      S += red[(w * MT + tid) * 2];
      Q += red[(w * MT + tid) * 2 + 1];
    }
    float mu  = S * (1.f / HID);
    float var = Q * (1.f / HID) - mu * mu;
    stats[tid]      = mu;
    stats[MT + tid] = rsqrtf(var + 1e-5f);
  }
  __syncthreads();

  // normalize + exact GELU -> bf16 activations in LDS
#pragma unroll
  for (int t = 0; t < 8; ++t) {
    int h = wave * 128 + t * 16 + nl;
    float g  = lng[e * HID + h];
    float bb = lnb[e * HID + h];
#pragma unroll
    for (int r = 0; r < 8; ++r) {
      int m0 = sub * 8 + r, m1 = 16 + sub * 8 + r;
      float h0 = (acc0[t][r] - stats[m0]) * stats[MT + m0] * g + bb;
      float h1 = (acc1[t][r] - stats[m1]) * stats[MT + m1] * g + bb;
      As[m0 * ASTR + h] = (bf16)(0.5f * h0 * (1.f + erff(h0 * 0.70710678118654752f)));
      As[m1 * ASTR + h] = (bf16)(0.5f * h1 * (1.f + erff(h1 * 0.70710678118654752f)));
    }
  }
  __syncthreads();

  // ---------------- GEMM2: y = As @ W2[e] ----------------
  v8f acc2[2][2];
#pragma unroll
  for (int mt = 0; mt < 2; ++mt)
#pragma unroll
    for (int t = 0; t < 2; ++t) acc2[mt][t] = v8f_zero();
  const bf16* AsA0 = As + nl * ASTR;
  const bf16* AsA1 = As + (16 + nl) * ASTR;
  for (int kk = 0; kk < HID; kk += 32) {
    FragAB a0, a1;
    a0.f4[0] = *reinterpret_cast<const f4v*>(AsA0 + kk + sub * 8);
    a0.f4[1] = *reinterpret_cast<const f4v*>(AsA0 + kk + 16 + sub * 8);
    a1.f4[0] = *reinterpret_cast<const f4v*>(AsA1 + kk + sub * 8);
    a1.f4[1] = *reinterpret_cast<const f4v*>(AsA1 + kk + 16 + sub * 8);
#pragma unroll
    for (int t = 0; t < 2; ++t) {
      int d = wave * 32 + t * 16 + nl;
      FragAB b;
      const f4v* bp = reinterpret_cast<const f4v*>(
          W2t + ((size_t)(e * DIM + d)) * HID + kk + sub * 8);
      b.f4[0] = bp[0]; b.f4[1] = bp[2];
      acc2[0][t] = __builtin_amdgcn_wmma_f32_16x16x32_bf16(
          false, a0.v, false, b.v, (short)0, acc2[0][t], false, false);
      acc2[1][t] = __builtin_amdgcn_wmma_f32_16x16x32_bf16(
          false, a1.v, false, b.v, (short)0, acc2[1][t], false, false);
    }
  }

  // scatter y + b2 into per-slot output rows (slot-unique -> no atomics)
#pragma unroll
  for (int t = 0; t < 2; ++t) {
    int d = wave * 32 + t * 16 + nl;
    float b2v = b2[e * DIM + d];
#pragma unroll
    for (int mt = 0; mt < 2; ++mt) {
#pragma unroll
      for (int r = 0; r < 8; ++r) {
        int m = mt * 16 + sub * 8 + r;
        if (m < valid) Ybuf[(size_t)ids[m] * DIM + d] = acc2[mt][t][r] + b2v;
      }
    }
  }
}

// ---------------------------------------------------------------------------
// Kernel 4: deterministic weighted combine + output LayerNorm.
// ---------------------------------------------------------------------------
__global__ void final_kernel(const float* __restrict__ Ybuf, const float* __restrict__ tw,
                             const float* __restrict__ outg, const float* __restrict__ outb,
                             float* __restrict__ out) {
  __shared__ float rs[8][2];
  __shared__ float st[2];
  int n = blockIdx.x, tid = threadIdx.x;
  float w0 = tw[2 * n], w1 = tw[2 * n + 1];
  const float* y0 = Ybuf + (size_t)(2 * n) * DIM;
  const float* y1 = y0 + DIM;
  int d0 = tid, d1 = tid + 256;
  float c0 = w0 * y0[d0] + w1 * y1[d0];
  float c1 = w0 * y0[d1] + w1 * y1[d1];
  float s = c0 + c1, q = c0 * c0 + c1 * c1;
#pragma unroll
  for (int m = 16; m >= 1; m >>= 1) { s += __shfl_xor(s, m, 32); q += __shfl_xor(q, m, 32); }
  int wave = tid >> 5, lane = tid & 31;
  if (lane == 0) { rs[wave][0] = s; rs[wave][1] = q; }
  __syncthreads();
  if (tid == 0) {
    float S = 0.f, Q = 0.f;
#pragma unroll
    for (int w = 0; w < 8; ++w) { S += rs[w][0]; Q += rs[w][1]; }
    float mu = S * (1.f / DIM);
    float var = Q * (1.f / DIM) - mu * mu;
    st[0] = mu; st[1] = rsqrtf(var + 1e-5f);
  }
  __syncthreads();
  float mu = st[0], r = st[1];
  out[(size_t)n * DIM + d0] = (c0 - mu) * r * outg[d0] + outb[d0];
  out[(size_t)n * DIM + d1] = (c1 - mu) * r * outg[d1] + outb[d1];
}

// ---------------------------------------------------------------------------
// Kernel 5: aux loss (fixed-order sums -> deterministic).
// ---------------------------------------------------------------------------
__global__ void aux_kernel(const float* __restrict__ imp_part, const int* __restrict__ cnt,
                           float* __restrict__ aux_out) {
  __shared__ float imp_s[8];
  int tid = threadIdx.x;
  if (tid < 8) {
    float s = 0.f;
#pragma unroll 8
    for (int b = 0; b < 1024; ++b) s += imp_part[b * 8 + tid];
    imp_s[tid] = s;
  }
  __syncthreads();
  if (tid == 0) {
    float imp[8], ld[8];
#pragma unroll
    for (int e = 0; e < 8; ++e) { imp[e] = imp_s[e]; ld[e] = (float)cnt[e] * (1.f / NTOK); }
    float mi = 0.f, ml = 0.f;
#pragma unroll
    for (int e = 0; e < 8; ++e) { mi += imp[e]; ml += ld[e]; }
    mi *= 0.125f; ml *= 0.125f;
    float vi = 0.f, vl = 0.f;
#pragma unroll
    for (int e = 0; e < 8; ++e) {
      float di = imp[e] - mi; vi += di * di;
      float dl = ld[e] - ml;  vl += dl * dl;
    }
    vi *= (1.f / 7.f); vl *= (1.f / 7.f);   // unbiased (ddof=1) variance
    float il = vi / ((mi + 1e-6f) * (mi + 1e-6f));
    float ll = vl / ((ml + 1e-6f) * (ml + 1e-6f));
    *aux_out = il + ll;
  }
}

// ---------------------------------------------------------------------------
extern "C" void kernel_launch(void* const* d_in, const int* in_sizes, int n_in,
                              void* d_out, int out_size, void* d_ws, size_t ws_size,
                              hipStream_t stream) {
  (void)in_sizes; (void)n_in; (void)out_size; (void)ws_size;
  const float* x   = (const float*)d_in[0];
  const float* gW  = (const float*)d_in[1];
  const float* gB  = (const float*)d_in[2];
  const float* W1  = (const float*)d_in[3];
  const float* b1  = (const float*)d_in[4];
  const float* lng = (const float*)d_in[5];
  const float* lnb = (const float*)d_in[6];
  const float* W2  = (const float*)d_in[7];
  const float* b2  = (const float*)d_in[8];
  const float* og  = (const float*)d_in[9];
  const float* ob  = (const float*)d_in[10];
  float* out = (float*)d_out;

  char* ws = (char*)d_ws;
  size_t off = 0;
  auto take = [&](size_t bytes) -> char* {
    char* p = ws + off;
    off = (off + bytes + 255) & ~(size_t)255;
    return p;
  };
  int*   cnt   = (int*)take(NE * sizeof(int));
  float* tw    = (float*)take((size_t)NTOK * 2 * sizeof(float));
  int*   tlist = (int*)take((size_t)NE * NTOK * sizeof(int));
  float* impp  = (float*)take((size_t)1024 * 8 * sizeof(float));
  bf16*  W1t   = (bf16*)take((size_t)NE * HID * DIM * sizeof(bf16));
  bf16*  W2t   = (bf16*)take((size_t)NE * HID * DIM * sizeof(bf16));
  float* Ybuf  = (float*)take((size_t)NTOK * 2 * DIM * sizeof(float));

  prep_kernel<<<32768, 256, 0, stream>>>(W1, W2, W1t, W2t, cnt);
  gate_kernel<<<1024, 256, 0, stream>>>(x, gW, gB, cnt, tlist, tw, impp);

  size_t lds = (size_t)32 * (DIM + 8) * 2     // Xs (bf16)
             + (size_t)32 * (HID + 8) * 2     // As (bf16) / fp32 x-staging
             + (size_t)16 * 32 * 2 * 4        // red
             + (size_t)2 * 32 * 4             // stats
             + (size_t)32 * 4;                // ids
  expert_kernel<<<dim3(256, NE), 512, lds, stream>>>(
      x, b1, lng, lnb, b2, cnt, tlist, W1t, W2t, Ybuf);

  final_kernel<<<NTOK, 256, 0, stream>>>(Ybuf, tw, og, ob, out);
  aux_kernel<<<1, 32, 0, stream>>>(impp, cnt, out + (size_t)NTOK * DIM);
}
